// Net_32787780337774
// MI455X (gfx1250) — matrix operand order repaired and checked by reading, not compiled, not run on hardware
//
#include <hip/hip_runtime.h>
#include <math.h>

// ---------------- constants (from reference) ----------------
#define Hh     128
#define Ww     128
#define NPIX   (Hh*Ww)          // 16384
#define Cch    32
#define NHEADS 8
#define Dd     4
#define Ll     128
#define HIDN   128              // 4*C
#define HID4   512              // 4*HIDN
#define ATTN_SCALE 0.17677669529663687f   // 32^-0.5

// ---------------- WMMA types ----------------
typedef _Float16 half16 __attribute__((ext_vector_type(16)));
typedef float    float8 __attribute__((ext_vector_type(8)));
typedef float    float2v __attribute__((ext_vector_type(2)));
union h16u { half16 v; _Float16 e[16]; };
union f8u  { float8 v; float e[8]; };
union f2u  { float2v v; float e[2]; };

#if defined(__HIP_DEVICE_COMPILE__) && __has_builtin(__builtin_amdgcn_wmma_f32_16x16x32_f16)
#define USE_WMMA_F16 1
#endif
#if defined(__HIP_DEVICE_COMPILE__) && __has_builtin(__builtin_amdgcn_wmma_f32_16x16x4_f32)
#define USE_WMMA_F32X4 1
#endif
// CDNA5 async global->LDS path (ASYNCcnt-tracked, 128b per lane per op)
// builtin signature (from clang diagnostic): (v4i32 AS1*, v4i32 AS3*, imm int, imm int)
#if defined(__HIP_DEVICE_COMPILE__) && \
    __has_builtin(__builtin_amdgcn_global_load_async_to_lds_b128) && \
    __has_builtin(__builtin_amdgcn_s_wait_asynccnt)
#define USE_ASYNC_LDS 1
typedef int v4i_t __attribute__((vector_size(16)));
typedef __attribute__((address_space(1))) v4i_t* gv4p;
typedef __attribute__((address_space(3))) v4i_t* lv4p;
#endif

__device__ __forceinline__ float gelu_f(float v) {
  return 0.5f * v * (1.0f + erff(v * 0.70710678118654752f));
}
// axial addressing: type 0 = "height" (fixed w=x, run over h=l); type 1 = "width"
__device__ __forceinline__ int axidx(int type, int l, int x) {
  return (type == 0) ? (l * Ww + x) : (x * Ww + l);
}

// ============================================================
// 1x1 conv as GEMM:  Y[Cout,P] = act(W[Cout,CIN] @ X[CIN,P] + b)
// one wave per 16(Cout) x 64(pixel) tile: A fragment loaded once,
// X tile staged in LDS via async global->LDS b128, 4 WMMAs / k-chunk.
// ACT: 0=none, 1=gelu
// ============================================================
template <int CIN, int ACT>
__global__ void k_pw_wmma(const float* __restrict__ X, const float* __restrict__ Wt,
                          const float* __restrict__ Bias, float* __restrict__ Y) {
  const int lane = threadIdx.x;
  const int p0  = blockIdx.x * 64;
  const int co0 = blockIdx.y * 16;
  const int m = lane & 15, g = lane >> 4;
  __shared__ __align__(16) float xs[32 * 64];   // K-chunk(32) x pixels(64), 8 KB

  f8u acc[4];
#pragma unroll
  for (int t = 0; t < 4; ++t)
#pragma unroll
    for (int r = 0; r < 8; ++r) acc[t].e[r] = 0.0f;

  for (int k0 = 0; k0 < CIN; k0 += 32) {
    // ---- stage X[k0+lane][p0 .. p0+63] into LDS row `lane` ----
    const float* src = &X[(size_t)(k0 + lane) * NPIX + p0];
#if USE_ASYNC_LDS
    gv4p gp = (gv4p)(unsigned long long)(const void*)src;      // AS1 value == generic value
    lv4p lp = (lv4p)(void*)(&xs[lane * 64]);                   // addrspacecast generic->AS3
#define ASYQ(q) __builtin_amdgcn_global_load_async_to_lds_b128(gp, lp, (q) * 16, 0)
    ASYQ(0);  ASYQ(1);  ASYQ(2);  ASYQ(3);
    ASYQ(4);  ASYQ(5);  ASYQ(6);  ASYQ(7);
    ASYQ(8);  ASYQ(9);  ASYQ(10); ASYQ(11);
    ASYQ(12); ASYQ(13); ASYQ(14); ASYQ(15);
#undef ASYQ
    __builtin_amdgcn_s_wait_asynccnt(0);
#else
    float4* dv = (float4*)&xs[lane * 64];
    const float4* sv = (const float4*)src;
#pragma unroll
    for (int q = 0; q < 16; ++q) dv[q] = sv[q];
#endif
    __syncthreads();

    // ---- A fragment (weights), loaded once per k-chunk ----
    h16u a;
#pragma unroll
    for (int h = 0; h < 16; ++h) {
      // A (16x32 f16): lane-group g holds K = g*8+h (h<8) and 16+g*8+(h-8)
      int ka = (h < 8) ? (g * 8 + h) : (16 + g * 8 + (h - 8));
      a.e[h] = (_Float16)Wt[(size_t)(co0 + m) * CIN + k0 + ka];
    }
#if USE_WMMA_F16
#pragma unroll
    for (int t = 0; t < 4; ++t) {
      h16u b;
#pragma unroll
      for (int h = 0; h < 16; ++h)   // B (32x16 f16): K = g*8+h (h<8) and 16+g*8+(h-8), N = lane&15
        b.e[h] = (_Float16)xs[((h < 8) ? (g * 8 + h) : (16 + g * 8 + (h - 8))) * 64 + t * 16 + m];
      acc[t].v = __builtin_amdgcn_wmma_f32_16x16x32_f16(false, a.v, false, b.v,
                                                        (short)0, acc[t].v, false, false);
    }
#else
#pragma unroll
    for (int t = 0; t < 4; ++t)
#pragma unroll
      for (int r = 0; r < 8; ++r) {
        int mm = r + 8 * g;
        float s = 0.f;
        for (int kk = 0; kk < 32; ++kk)
          s += Wt[(size_t)(co0 + mm) * CIN + k0 + kk] * xs[kk * 64 + t * 16 + m];
        acc[t].e[r] += s;
      }
#endif
    __syncthreads();
  }
#pragma unroll
  for (int t = 0; t < 4; ++t)
#pragma unroll
    for (int r = 0; r < 8; ++r) {
      int mm = r + 8 * g;               // D: vgpr r, lane-group g -> row r+8g
      float vv = acc[t].e[r] + Bias[co0 + mm];
      if (ACT == 1) vv = gelu_f(vv);
      Y[(size_t)(co0 + mm) * NPIX + p0 + t * 16 + m] = vv;
    }
}

// ============================================================
// axial QK^T (+scale) + row softmax -> attn[(bx, head, L, L)]
// one wave per (x, head, 16-row strip); K=4 exact fp32 WMMA
// ============================================================
__global__ void k_axial_qk_softmax(const float* __restrict__ q, const float* __restrict__ kf,
                                   float* __restrict__ attn, int type) {
  const int lane = threadIdx.x;
  const int i0 = blockIdx.x * 16;
  const int n  = blockIdx.y;
  const int x  = blockIdx.z;
  const int m = lane & 15, g = lane >> 4;
  __shared__ float s[16 * Ll];

  f2u a;  // A (16x4 f32): vgpr r holds K = 2g + r, row = lane&15
  a.e[0] = q[(size_t)(n * Dd + (g * 2 + 0)) * NPIX + axidx(type, i0 + m, x)];
  a.e[1] = q[(size_t)(n * Dd + (g * 2 + 1)) * NPIX + axidx(type, i0 + m, x)];

  for (int jt = 0; jt < 8; ++jt) {
    const int j0 = jt * 16;
    f2u b;  // B (4x16 f32): mirrored pattern, col = lane&15
    b.e[0] = kf[(size_t)(n * Dd + (g * 2 + 0)) * NPIX + axidx(type, j0 + m, x)];
    b.e[1] = kf[(size_t)(n * Dd + (g * 2 + 1)) * NPIX + axidx(type, j0 + m, x)];
    f8u c;
#pragma unroll
    for (int r = 0; r < 8; ++r) c.e[r] = 0.0f;
#if USE_WMMA_F32X4
    c.v = __builtin_amdgcn_wmma_f32_16x16x4_f32(false, a.v, false, b.v,
                                                (short)0, c.v, false, false);
#else
#pragma unroll
    for (int r = 0; r < 8; ++r) {
      int mm = r + 8 * g;
      float sacc = 0.f;
      for (int dd = 0; dd < Dd; ++dd)
        sacc += q[(size_t)(n * Dd + dd) * NPIX + axidx(type, i0 + mm, x)] *
                kf[(size_t)(n * Dd + dd) * NPIX + axidx(type, j0 + m, x)];
      c.e[r] = sacc;
    }
#endif
#pragma unroll
    for (int r = 0; r < 8; ++r)
      s[(r + 8 * g) * Ll + j0 + m] = c.e[r] * ATTN_SCALE;
  }
  __syncthreads();
  if (lane < 16) {
    float* row = &s[lane * Ll];
    float mx = -1e30f;
    for (int j = 0; j < Ll; ++j) mx = fmaxf(mx, row[j]);
    float sum = 0.f;
    for (int j = 0; j < Ll; ++j) { float e = expf(row[j] - mx); row[j] = e; sum += e; }
    float inv = 1.0f / sum;
    float* dst = &attn[(((size_t)x * NHEADS + n) * Ll + (i0 + lane)) * Ll];
    for (int j = 0; j < Ll; ++j) dst[j] = row[j] * inv;
  }
}

// ============================================================
// O = attn @ V, written back into NCHW layout (pre-o-projection)
// ============================================================
__global__ void k_av(const float* __restrict__ attn, const float* __restrict__ v,
                     float* __restrict__ out, int type) {
  const int n = blockIdx.x, bx = blockIdx.y, i = threadIdx.x;
  const float* arow = &attn[(((size_t)bx * NHEADS + n) * Ll + i) * Ll];
  float acc[Dd] = {0.f, 0.f, 0.f, 0.f};
  for (int j = 0; j < Ll; ++j) {
    float aij = arow[j];
    int off = axidx(type, j, bx);
#pragma unroll
    for (int dd = 0; dd < Dd; ++dd)
      acc[dd] += aij * v[(size_t)(n * Dd + dd) * NPIX + off];
  }
  int oo = axidx(type, i, bx);
#pragma unroll
  for (int dd = 0; dd < Dd; ++dd)
    out[(size_t)(n * Dd + dd) * NPIX + oo] = acc[dd];
}

// ============================================================
// generic 3x3 conv (SAME), optional relu + residual add
// ============================================================
__global__ void k_conv3x3(const float* __restrict__ x, const float* __restrict__ w,
                          const float* __restrict__ b, float* __restrict__ y,
                          int Cin, int relu, const float* __restrict__ resid) {
  const int p = blockIdx.x * 256 + threadIdx.x;
  const int co = blockIdx.y;
  const int i = p >> 7, j = p & (Ww - 1);
  float acc = b[co];
  for (int ci = 0; ci < Cin; ++ci) {
    const float* xp = &x[(size_t)ci * NPIX];
    const float* wp = &w[((size_t)co * Cin + ci) * 9];
#pragma unroll
    for (int di = 0; di < 3; ++di) {
      int yy = i + di - 1; if (yy < 0 || yy >= Hh) continue;
#pragma unroll
      for (int dj = 0; dj < 3; ++dj) {
        int xx = j + dj - 1; if (xx < 0 || xx >= Ww) continue;
        acc += xp[yy * Ww + xx] * wp[di * 3 + dj];
      }
    }
  }
  if (relu) acc = fmaxf(acc, 0.f);
  if (resid) acc += resid[(size_t)co * NPIX + p];
  y[(size_t)co * NPIX + p] = acc;
}

// ============================================================
// channel layernorm (over C=32 of NCHW), eps=1e-5, biased var
// ============================================================
__global__ void k_ln_c(const float* __restrict__ x, const float* __restrict__ g,
                       const float* __restrict__ b, float* __restrict__ y) {
  const int p = blockIdx.x * 256 + threadIdx.x;
  float m = 0.f;
  for (int c = 0; c < Cch; ++c) m += x[(size_t)c * NPIX + p];
  m *= (1.0f / Cch);
  float v = 0.f;
  for (int c = 0; c < Cch; ++c) { float d = x[(size_t)c * NPIX + p] - m; v += d * d; }
  v *= (1.0f / Cch);
  float inv = rsqrtf(v + 1e-5f);
  for (int c = 0; c < Cch; ++c)
    y[(size_t)c * NPIX + p] = (x[(size_t)c * NPIX + p] - m) * inv * g[c] + b[c];
}

// concat([base+oh, base+ow]) -> 64ch
__global__ void k_concat_add(const float* __restrict__ base, const float* __restrict__ oh,
                             const float* __restrict__ ow, float* __restrict__ y) {
  const int p = blockIdx.x * 256 + threadIdx.x;
  const int c = blockIdx.y;
  float r = (c < Cch) ? (base[(size_t)c * NPIX + p] + oh[(size_t)c * NPIX + p])
                      : (base[(size_t)(c - Cch) * NPIX + p] + ow[(size_t)(c - Cch) * NPIX + p]);
  y[(size_t)c * NPIX + p] = r;
}

__global__ void k_add(const float* __restrict__ a, const float* __restrict__ b,
                      float* __restrict__ y, int n) {
  int i = blockIdx.x * 256 + threadIdx.x;
  if (i < n) y[i] = a[i] + b[i];
}

// depthwise KSxKS conv on 128-ch hidden, gelu at write (gelu over concat)
template <int KS>
__global__ void k_dw_gelu(const float* __restrict__ x, const float* __restrict__ w,
                          const float* __restrict__ b, float* __restrict__ y, int chOff) {
  const int p = blockIdx.x * 256 + threadIdx.x;
  const int ch = blockIdx.y;
  const int i = p >> 7, j = p & (Ww - 1);
  const int pad = KS >> 1;
  float acc = b[ch];
#pragma unroll
  for (int di = 0; di < KS; ++di) {
    int yy = i + di - pad; if (yy < 0 || yy >= Hh) continue;
#pragma unroll
    for (int dj = 0; dj < KS; ++dj) {
      int xx = j + dj - pad; if (xx < 0 || xx >= Ww) continue;
      acc += x[(size_t)ch * NPIX + yy * Ww + xx] * w[(size_t)ch * KS * KS + di * KS + dj];
    }
  }
  y[(size_t)(chOff + ch) * NPIX + p] = gelu_f(acc);
}

// ============================================================
// adaptive kernel: fused conv1 (heads->3, (1,3), pad j) + relu
//                  + grouped 3x3 (3->3) + relu  => a2[bx,3,L,L]
// ============================================================
__global__ void k_ak_a2(const float* __restrict__ attn,
                        const float* __restrict__ c1w, const float* __restrict__ c1b,
                        const float* __restrict__ c2w, const float* __restrict__ c2b,
                        float* __restrict__ a2) {
  const int p = blockIdx.x * 256 + threadIdx.x;
  const int kk = blockIdx.y;
  const int bx = blockIdx.z;
  const int i = p >> 7, j = p & (Ll - 1);
  const float* A = &attn[(size_t)bx * NHEADS * Ll * Ll];
  float acc = c2b[kk];
  for (int di = 0; di < 3; ++di) {
    int yy = i + di - 1; if (yy < 0 || yy >= Ll) continue;
    for (int dj = 0; dj < 3; ++dj) {
      int xj = j + dj - 1; if (xj < 0 || xj >= Ll) continue;
      float a1 = c1b[kk];
      for (int nh = 0; nh < NHEADS; ++nh) {
        const float* rowp = &A[((size_t)nh * Ll + yy) * Ll];
#pragma unroll
        for (int t = 0; t < 3; ++t) {
          int jj = xj + t - 1; if (jj < 0 || jj >= Ll) continue;
          a1 += rowp[jj] * c1w[(kk * NHEADS + nh) * 3 + t];
        }
      }
      a1 = fmaxf(a1, 0.f);
      acc += a1 * c2w[kk * 9 + di * 3 + dj];
    }
  }
  a2[(((size_t)bx * 3 + kk) * Ll * Ll) + p] = fmaxf(acc, 0.f);
}

// pooled = mean(v over L,d); val[bx,n,kk] = pooled*lw[kk]+lb[kk]
__global__ void k_pool_val(const float* __restrict__ v, const float* __restrict__ lw,
                           const float* __restrict__ lb, float* __restrict__ val, int type) {
  __shared__ float red[Ll];
  const int n = blockIdx.x, bx = blockIdx.y, l = threadIdx.x;
  float s = 0.f;
  int off = axidx(type, l, bx);
#pragma unroll
  for (int dd = 0; dd < Dd; ++dd) s += v[(size_t)(n * Dd + dd) * NPIX + off];
  red[l] = s;
  __syncthreads();
  for (int st = 64; st > 0; st >>= 1) { if (l < st) red[l] += red[l + st]; __syncthreads(); }
  if (l == 0) {
    float pooled = red[0] / (float)(Ll * Dd);
    for (int kk = 0; kk < 3; ++kk)
      val[((size_t)bx * NHEADS + n) * 3 + kk] = pooled * lw[kk] + lb[kk];
  }
}

// fused adaptive_conv (dynamic 1D conv along j) + attn@v -> NCHW
__global__ void k_adaptive_av(const float* __restrict__ attn, const float* __restrict__ a2,
                              const float* __restrict__ val, const float* __restrict__ v,
                              float* __restrict__ out, int type) {
  const int n = blockIdx.x, bx = blockIdx.y, i = threadIdx.x;
  const float* arow = &attn[(((size_t)bx * NHEADS + n) * Ll + i) * Ll];
  const float* a2b  = &a2[((size_t)bx * 3) * Ll * Ll + (size_t)i * Ll];
  const float v0 = val[((size_t)bx * NHEADS + n) * 3 + 0];
  const float v1 = val[((size_t)bx * NHEADS + n) * 3 + 1];
  const float v2 = val[((size_t)bx * NHEADS + n) * 3 + 2];
  float acc[Dd] = {0.f, 0.f, 0.f, 0.f};
  for (int j = 0; j < Ll; ++j) {
    float na = arow[j] * (a2b[(size_t)1 * Ll * Ll + j] * v1);
    if (j > 0)      na += arow[j - 1] * (a2b[j] * v0);
    if (j < Ll - 1) na += arow[j + 1] * (a2b[(size_t)2 * Ll * Ll + j] * v2);
    int off = axidx(type, j, bx);
#pragma unroll
    for (int dd = 0; dd < Dd; ++dd)
      acc[dd] += na * v[(size_t)(n * Dd + dd) * NPIX + off];
  }
  int oo = axidx(type, i, bx);
#pragma unroll
  for (int dd = 0; dd < Dd; ++dd)
    out[(size_t)(n * Dd + dd) * NPIX + oo] = acc[dd];
}

// ============================================================
// host orchestration
// ============================================================
enum {
  IN_LMS = 0, IN_PAN,
  B1_AH_KB, B1_AH_KW, B1_AH_OB, B1_AH_OW, B1_AH_QB, B1_AH_QW, B1_AH_VB, B1_AH_VW,
  B1_AW_KB, B1_AW_KW, B1_AW_OB, B1_AW_OW, B1_AW_QB, B1_AW_QW, B1_AW_VB, B1_AW_VW,
  B1_CB, B1_CW,
  B1_M_B1, B1_M_B3, B1_M_DB1, B1_M_DB3, B1_M_DB5, B1_M_DB7,
  B1_M_DW1, B1_M_DW3, B1_M_DW5, B1_M_DW7, B1_M_W1, B1_M_W3,
  B1_N1B, B1_N1G, B1_N2B, B1_N2G, B1_N3B, B1_N3G,
  B2_AH_C1B, B2_AH_C1W, B2_AH_C2B, B2_AH_C2W, B2_AH_LB, B2_AH_LW,
  B2_AH_OB, B2_AH_OW, B2_AH_VB, B2_AH_VW,
  B2_AW_C1B, B2_AW_C1W, B2_AW_C2B, B2_AW_C2W, B2_AW_LB, B2_AW_LW,
  B2_AW_OB, B2_AW_OW, B2_AW_VB, B2_AW_VW,
  B2_CB, B2_CW,
  B2_M_B1, B2_M_B3, B2_M_DB1, B2_M_DB3, B2_M_DB5, B2_M_DB7,
  B2_M_DW1, B2_M_DW3, B2_M_DW5, B2_M_DW7, B2_M_W1, B2_M_W3,
  B2_N1B, B2_N1G, B2_N3B, B2_N3G,
  OUT_B1, OUT_B2, OUT_W1, OUT_W2,
  PL_B1, PL_B2, PL_W1, PL_W2,
  PPN_B1, PPN_B2, PPN_W1, PPN_W2,
  N_PARAMS
};

extern "C" void kernel_launch(void* const* d_in, const int* in_sizes, int n_in,
                              void* d_out, int out_size, void* d_ws, size_t ws_size,
                              hipStream_t stream) {
  (void)in_sizes; (void)n_in; (void)out_size; (void)ws_size;
  const float* A[N_PARAMS];
  for (int i = 0; i < N_PARAMS; ++i) A[i] = (const float*)d_in[i];
  const float* lms = A[IN_LMS];
  const float* pan = A[IN_PAN];

  // ---- workspace carve-up (f32 elements) ----
  float* base = (float*)d_ws;
  size_t off = 0;
  auto alloc = [&](size_t n) { float* p = base + off; off += n; return p; };
  float* T0  = alloc((size_t)Cch * NPIX);
  float* PF  = alloc((size_t)Cch * NPIX);
  float* LF  = alloc((size_t)Cch * NPIX);
  float* PN  = alloc((size_t)Cch * NPIX);
  float* LNb = alloc((size_t)Cch * NPIX);
  float* QB  = alloc((size_t)Cch * NPIX);
  float* KB  = alloc((size_t)Cch * NPIX);
  float* VB  = alloc((size_t)Cch * NPIX);
  float* AV  = alloc((size_t)Cch * NPIX);
  float* OHb = alloc((size_t)Cch * NPIX);
  float* OWb = alloc((size_t)Cch * NPIX);
  float* CCb = alloc((size_t)2 * Cch * NPIX);
  float* X1  = alloc((size_t)Cch * NPIX);
  float* X2  = alloc((size_t)Cch * NPIX);
  float* NBb = alloc((size_t)Cch * NPIX);
  float* MOb = alloc((size_t)Cch * NPIX);
  float* HIDb= alloc((size_t)HIDN * NPIX);
  float* H4b = alloc((size_t)HID4 * NPIX);
  float* AHp = alloc((size_t)Ww * NHEADS * Ll * Ll);
  float* AWp = alloc((size_t)Hh * NHEADS * Ll * Ll);
  float* A2p = alloc((size_t)128 * 3 * Ll * Ll);
  float* VALp= alloc((size_t)128 * NHEADS * 3);
  float* T8  = alloc((size_t)8 * NPIX);

  const dim3 B256(256);
  auto conv3 = [&](const float* x, int iw, int ib, float* y, int Cin, int Cout,
                   int relu, const float* resid) {
    k_conv3x3<<<dim3(NPIX / 256, Cout), B256, 0, stream>>>(x, A[iw], A[ib], y, Cin, relu, resid);
  };
  auto pw = [&](const float* x, int iw, int ib, float* y, int Cin, int Cout, int act) {
    dim3 grid(NPIX / 64, Cout / 16);
    if (Cin == Cch && act == 0)
      k_pw_wmma<Cch, 0><<<grid, 32, 0, stream>>>(x, A[iw], A[ib], y);
    else if (Cin == Cch && act == 1)
      k_pw_wmma<Cch, 1><<<grid, 32, 0, stream>>>(x, A[iw], A[ib], y);
    else
      k_pw_wmma<HID4, 0><<<grid, 32, 0, stream>>>(x, A[iw], A[ib], y);
  };
  auto lnrm = [&](const float* x, int ig, int ib, float* y) {
    k_ln_c<<<dim3(NPIX / 256), B256, 0, stream>>>(x, A[ig], A[ib], y);
  };
  auto mlp = [&](const float* xin, float* xout, int W1, int Bi1, int DW1, int DB1,
                 int DW3, int DB3, int DW5, int DB5, int DW7, int DB7, int W3, int Bi3) {
    pw(xin, W1, Bi1, HIDb, Cch, HIDN, 1);  // gelu(conv1x1 32->128)
    k_dw_gelu<1><<<dim3(NPIX / 256, HIDN), B256, 0, stream>>>(HIDb, A[DW1], A[DB1], H4b, 0);
    k_dw_gelu<3><<<dim3(NPIX / 256, HIDN), B256, 0, stream>>>(HIDb, A[DW3], A[DB3], H4b, 128);
    k_dw_gelu<5><<<dim3(NPIX / 256, HIDN), B256, 0, stream>>>(HIDb, A[DW5], A[DB5], H4b, 256);
    k_dw_gelu<7><<<dim3(NPIX / 256, HIDN), B256, 0, stream>>>(HIDb, A[DW7], A[DB7], H4b, 384);
    pw(H4b, W3, Bi3, xout, HID4, Cch, 0);
  };

  // ---------------- projections ----------------
  conv3(pan, PPN_W1, PPN_B1, T0, 1, Cch, 1, nullptr);
  conv3(T0, PPN_W2, PPN_B2, PF, Cch, Cch, 0, nullptr);
  conv3(lms, PL_W1, PL_B1, T0, 8, Cch, 1, nullptr);
  conv3(T0, PL_W2, PL_B2, LF, Cch, Cch, 0, nullptr);

  // ---------------- block1 ----------------
  lnrm(PF, B1_N1G, B1_N1B, PN);
  lnrm(LF, B1_N2G, B1_N2B, LNb);
  // height axial attention
  pw(PN, B1_AH_QW, B1_AH_QB, QB, Cch, Cch, 0);
  pw(LNb, B1_AH_KW, B1_AH_KB, KB, Cch, Cch, 0);
  pw(LNb, B1_AH_VW, B1_AH_VB, VB, Cch, Cch, 0);
  k_axial_qk_softmax<<<dim3(8, NHEADS, Ww), 32, 0, stream>>>(QB, KB, AHp, 0);
  k_av<<<dim3(NHEADS, Ww), 128, 0, stream>>>(AHp, VB, AV, 0);
  pw(AV, B1_AH_OW, B1_AH_OB, OHb, Cch, Cch, 0);
  // width axial attention
  pw(PN, B1_AW_QW, B1_AW_QB, QB, Cch, Cch, 0);
  pw(LNb, B1_AW_KW, B1_AW_KB, KB, Cch, Cch, 0);
  pw(LNb, B1_AW_VW, B1_AW_VB, VB, Cch, Cch, 0);
  k_axial_qk_softmax<<<dim3(8, NHEADS, Hh), 32, 0, stream>>>(QB, KB, AWp, 1);
  k_av<<<dim3(NHEADS, Hh), 128, 0, stream>>>(AWp, VB, AV, 1);
  pw(AV, B1_AW_OW, B1_AW_OB, OWb, Cch, Cch, 0);
  // merge + mlp
  k_concat_add<<<dim3(NPIX / 256, 2 * Cch), B256, 0, stream>>>(LF, OHb, OWb, CCb);
  conv3(CCb, B1_CW, B1_CB, X1, 2 * Cch, Cch, 0, nullptr);
  lnrm(X1, B1_N3G, B1_N3B, NBb);
  mlp(NBb, MOb, B1_M_W1, B1_M_B1, B1_M_DW1, B1_M_DB1, B1_M_DW3, B1_M_DB3,
      B1_M_DW5, B1_M_DB5, B1_M_DW7, B1_M_DB7, B1_M_W3, B1_M_B3);
  k_add<<<dim3((Cch * NPIX) / 256), B256, 0, stream>>>(X1, MOb, X2, Cch * NPIX);

  // ---------------- block2 (adaptive attention) ----------------
  lnrm(X2, B2_N1G, B2_N1B, NBb);
  // height
  pw(NBb, B2_AH_VW, B2_AH_VB, VB, Cch, Cch, 0);
  k_ak_a2<<<dim3(NPIX / 256, 3, Ww), B256, 0, stream>>>(AHp, A[B2_AH_C1W], A[B2_AH_C1B],
                                                        A[B2_AH_C2W], A[B2_AH_C2B], A2p);
  k_pool_val<<<dim3(NHEADS, Ww), 128, 0, stream>>>(VB, A[B2_AH_LW], A[B2_AH_LB], VALp, 0);
  k_adaptive_av<<<dim3(NHEADS, Ww), 128, 0, stream>>>(AHp, A2p, VALp, VB, AV, 0);
  pw(AV, B2_AH_OW, B2_AH_OB, OHb, Cch, Cch, 0);
  // width
  pw(NBb, B2_AW_VW, B2_AW_VB, VB, Cch, Cch, 0);
  k_ak_a2<<<dim3(NPIX / 256, 3, Hh), B256, 0, stream>>>(AWp, A[B2_AW_C1W], A[B2_AW_C1B],
                                                        A[B2_AW_C2W], A[B2_AW_C2B], A2p);
  k_pool_val<<<dim3(NHEADS, Hh), 128, 0, stream>>>(VB, A[B2_AW_LW], A[B2_AW_LB], VALp, 1);
  k_adaptive_av<<<dim3(NHEADS, Hh), 128, 0, stream>>>(AWp, A2p, VALp, VB, AV, 1);
  pw(AV, B2_AW_OW, B2_AW_OB, OWb, Cch, Cch, 0);
  // merge + mlp
  k_concat_add<<<dim3(NPIX / 256, 2 * Cch), B256, 0, stream>>>(X2, OHb, OWb, CCb);
  conv3(CCb, B2_CW, B2_CB, X1, 2 * Cch, Cch, 0, nullptr);
  lnrm(X1, B2_N3G, B2_N3B, NBb);
  mlp(NBb, MOb, B2_M_W1, B2_M_B1, B2_M_DW1, B2_M_DB1, B2_M_DW3, B2_M_DB3,
      B2_M_DW5, B2_M_DB5, B2_M_DW7, B2_M_DB7, B2_M_W3, B2_M_B3);
  k_add<<<dim3((Cch * NPIX) / 256), B256, 0, stream>>>(X1, MOb, X2, Cch * NPIX);

  // ---------------- output projection + residual lms ----------------
  conv3(X2, OUT_W1, OUT_B1, T8, Cch, 8, 1, nullptr);
  k_conv3x3<<<dim3(NPIX / 256, 8), B256, 0, stream>>>(T8, A[OUT_W2], A[OUT_B2],
                                                      (float*)d_out, 8, 0, lms);
}
